// C3ALinear_77644418777897
// MI455X (gfx1250) — matrix operand order repaired and checked by reading, not compiled, not run on hardware
//
#include <hip/hip_runtime.h>

// Problem sizes (fixed by the reference)
#define M_TOTAL 8192   // B*S = 4*2048
#define N_TOTAL 4096   // D_OUT
#define K_TOTAL 4096   // D_IN

// GEMM tiling
#define BM 128
#define BN 128
#define BK 32
#define LDS_STRIDE 48            // 32 data + 16 pad elems: 96B rows, 16B-aligned chunks
#define STAGE_ELEMS (BM * LDS_STRIDE)

typedef __attribute__((ext_vector_type(16))) __bf16 v16bf;
typedef __attribute__((ext_vector_type(8)))  __bf16 v8bf;
typedef __attribute__((ext_vector_type(8)))  float  v8f;

// ---- CDNA5 async global->LDS path -----------------------------------------
#if defined(__has_builtin)
#if __has_builtin(__builtin_amdgcn_global_load_async_to_lds_b128)
#define USE_ASYNC_LDS 1
#endif
#endif

#if defined(USE_ASYNC_LDS)
// Builtin signature (from the round-2 diagnostic): param0 is
// 'int __attribute__((vector_size(16))) __device__ *'  == addrspace(1) v4i*.
typedef int v4i_vs __attribute__((vector_size(16)));
typedef __attribute__((address_space(1))) v4i_vs* glb_v4i_p;
typedef __attribute__((address_space(3))) v4i_vs* lds_v4i_p;
#define ASYNC_B128(ldsp, gp)                                        \
  __builtin_amdgcn_global_load_async_to_lds_b128(                   \
      (glb_v4i_p)(unsigned long long)(gp),                          \
      (lds_v4i_p)(unsigned int)(unsigned long long)(ldsp), 0, 0)
#if __has_builtin(__builtin_amdgcn_s_wait_asynccnt)
#define WAIT_ASYNC(n) __builtin_amdgcn_s_wait_asynccnt(n)
#else
#define WAIT_ASYNC(n) asm volatile("s_wait_asynccnt %0" ::"i"(n) : "memory")
#endif
#endif

__device__ __forceinline__ unsigned short f32_to_bf16_rne(float f) {
  unsigned u = __float_as_uint(f);
  unsigned r = u + 0x7FFFu + ((u >> 16) & 1u);  // round to nearest even
  return (unsigned short)(r >> 16);
}
__device__ __forceinline__ float bf16_bits_to_f32(unsigned short h) {
  return __uint_as_float(((unsigned)h) << 16);
}

// ---------------------------------------------------------------------------
// Kernel 1: split activations x (fp32) into bf16 hi/lo pair (bf16x3 scheme)
// ---------------------------------------------------------------------------
__global__ __launch_bounds__(256) void c3a_split_x(const float* __restrict__ x,
                                                   unsigned short* __restrict__ xhi,
                                                   unsigned short* __restrict__ xlo) {
  size_t i = (size_t)blockIdx.x * blockDim.x + threadIdx.x;
  float v = x[i];
  unsigned short h = f32_to_bf16_rne(v);
  xhi[i] = h;
  xlo[i] = f32_to_bf16_rne(v - bf16_bits_to_f32(h));
}

// ---------------------------------------------------------------------------
// Kernel 2: W_eff = base_weight + circulant(c3a_kernel)/D_IN, split to bf16 hi/lo
// ---------------------------------------------------------------------------
__global__ __launch_bounds__(256) void c3a_build_w(const float* __restrict__ w,
                                                   const float* __restrict__ ck,
                                                   unsigned short* __restrict__ whi,
                                                   unsigned short* __restrict__ wlo) {
  size_t idx = (size_t)blockIdx.x * blockDim.x + threadIdx.x;  // 0 .. 16M-1
  int o = (int)(idx >> 12);       // output feature (row, N)
  int k = (int)(idx & 4095);      // input feature (col, K)
  int p = o >> 8, i = o & 255;
  int q = k >> 8, j = k & 255;
  float val = w[idx] + ck[(((p << 4) + q) << 8) + ((i - j) & 255)] * (1.0f / 4096.0f);
  unsigned short h = f32_to_bf16_rne(val);
  whi[idx] = h;
  wlo[idx] = f32_to_bf16_rne(val - bf16_bits_to_f32(h));
}

// ---------------------------------------------------------------------------
// Kernel 3: GEMM  out[M][N] = Xeff @ Weff^T + bias   (bf16x3 via WMMA)
// 256 threads = 8 wave32 in a 4(M) x 2(N) grid; each wave owns a 32x64 slab
// of the 128x128 block tile -> 8 C tiles, 24 WMMAs per k-step per wave.
// Double-buffered LDS; async global->LDS when the toolchain exposes it.
// ---------------------------------------------------------------------------
__global__ __launch_bounds__(256) void c3a_gemm(const unsigned short* __restrict__ Ahi_g,
                                                const unsigned short* __restrict__ Alo_g,
                                                const unsigned short* __restrict__ Bhi_g,
                                                const unsigned short* __restrict__ Blo_g,
                                                const float* __restrict__ bias,
                                                float* __restrict__ out) {
  __shared__ unsigned short sAhi[2][STAGE_ELEMS];
  __shared__ unsigned short sAlo[2][STAGE_ELEMS];
  __shared__ unsigned short sBhi[2][STAGE_ELEMS];
  __shared__ unsigned short sBlo[2][STAGE_ELEMS];

  const int tid  = threadIdx.x;
  const int lane = tid & 31;
  const int wave = tid >> 5;     // 0..7
  const int wm   = wave >> 1;    // 0..3 -> M slab
  const int wn   = wave & 1;     // 0..1 -> N slab
  const int half = lane >> 4;    // which K-half this lane holds
  const int l16  = lane & 15;

  const int mBase = blockIdx.y * BM;
  const int nBase = blockIdx.x * BN;

  // Staging: each thread copies one 32B span (two 16B chunks) per matrix/stage.
  const int srow = tid >> 1;           // 0..127
  const int scol = (tid & 1) << 4;     // 0 or 16 elements

  // Stage global -> LDS for buffer `buf`, k-offset k0 (8 x 16B per thread).
  auto stage = [&](int buf, int k0) {
    size_t gA = (size_t)(mBase + srow) * K_TOTAL + k0 + scol;
    size_t gB = (size_t)(nBase + srow) * K_TOTAL + k0 + scol;
    int l = srow * LDS_STRIDE + scol;
#if defined(USE_ASYNC_LDS)
    ASYNC_B128(&sAhi[buf][l],     &Ahi_g[gA]);
    ASYNC_B128(&sAhi[buf][l + 8], &Ahi_g[gA + 8]);
    ASYNC_B128(&sAlo[buf][l],     &Alo_g[gA]);
    ASYNC_B128(&sAlo[buf][l + 8], &Alo_g[gA + 8]);
    ASYNC_B128(&sBhi[buf][l],     &Bhi_g[gB]);
    ASYNC_B128(&sBhi[buf][l + 8], &Bhi_g[gB + 8]);
    ASYNC_B128(&sBlo[buf][l],     &Blo_g[gB]);
    ASYNC_B128(&sBlo[buf][l + 8], &Blo_g[gB + 8]);
#else
    *(uint4*)&sAhi[buf][l]     = *(const uint4*)&Ahi_g[gA];
    *(uint4*)&sAhi[buf][l + 8] = *(const uint4*)&Ahi_g[gA + 8];
    *(uint4*)&sAlo[buf][l]     = *(const uint4*)&Alo_g[gA];
    *(uint4*)&sAlo[buf][l + 8] = *(const uint4*)&Alo_g[gA + 8];
    *(uint4*)&sBhi[buf][l]     = *(const uint4*)&Bhi_g[gB];
    *(uint4*)&sBhi[buf][l + 8] = *(const uint4*)&Bhi_g[gB + 8];
    *(uint4*)&sBlo[buf][l]     = *(const uint4*)&Blo_g[gB];
    *(uint4*)&sBlo[buf][l + 8] = *(const uint4*)&Blo_g[gB + 8];
#endif
  };

  // Accumulators: 8 tiles of 16x16 fp32. C/D layout: elem r of lane L is
  // row M = r + 8*(L>>4), col N = L&15. Bias uniform per column.
  v8f acc[8];
#pragma unroll
  for (int nt = 0; nt < 4; ++nt) {
    float bv = bias[nBase + wn * 64 + nt * 16 + l16];
#pragma unroll
    for (int mt = 0; mt < 2; ++mt)
#pragma unroll
      for (int r = 0; r < 8; ++r) acc[mt * 4 + nt][r] = bv;
  }

  stage(0, 0);  // prologue

  const int NSTEP = K_TOTAL / BK;  // 128
  for (int it = 0; it < NSTEP; ++it) {
    const int cur = it & 1;
    if (it + 1 < NSTEP) {
      stage(cur ^ 1, (it + 1) * BK);
#if defined(USE_ASYNC_LDS)
      WAIT_ASYNC(8);   // 8 just-issued may be in flight; current stage complete
#endif
    } else {
#if defined(USE_ASYNC_LDS)
      WAIT_ASYNC(0);
#endif
    }
    __syncthreads();   // all threads' stage data visible in LDS

#pragma unroll
    for (int mt = 0; mt < 2; ++mt) {
      // A fragment (16x32 bf16): lane holds row M=l16; elems 0..7 = K half*8..+7,
      // elems 8..15 = K 16+half*8..+7 (ISA 16-bit A layout).
      const int arow = wm * 32 + mt * 16 + l16;
      v8bf a0h = *(const v8bf*)&sAhi[cur][arow * LDS_STRIDE + half * 8];
      v8bf a1h = *(const v8bf*)&sAhi[cur][arow * LDS_STRIDE + 16 + half * 8];
      v8bf a0l = *(const v8bf*)&sAlo[cur][arow * LDS_STRIDE + half * 8];
      v8bf a1l = *(const v8bf*)&sAlo[cur][arow * LDS_STRIDE + 16 + half * 8];
      v16bf ah = __builtin_shufflevector(a0h, a1h, 0,1,2,3,4,5,6,7,8,9,10,11,12,13,14,15);
      v16bf al = __builtin_shufflevector(a0l, a1l, 0,1,2,3,4,5,6,7,8,9,10,11,12,13,14,15);

#pragma unroll
      for (int nt = 0; nt < 4; ++nt) {
        // B fragment (32x16 bf16): lane holds col N=l16; elems 0..7 = K half*8..+7,
        // elems 8..15 = K 16+half*8..+7 (same as A).
        const int brow = wn * 64 + nt * 16 + l16;
        v8bf b0h = *(const v8bf*)&sBhi[cur][brow * LDS_STRIDE + half * 8];
        v8bf b1h = *(const v8bf*)&sBhi[cur][brow * LDS_STRIDE + 16 + half * 8];
        v8bf b0l = *(const v8bf*)&sBlo[cur][brow * LDS_STRIDE + half * 8];
        v8bf b1l = *(const v8bf*)&sBlo[cur][brow * LDS_STRIDE + 16 + half * 8];
        v16bf bh = __builtin_shufflevector(b0h, b1h, 0,1,2,3,4,5,6,7,8,9,10,11,12,13,14,15);
        v16bf bl = __builtin_shufflevector(b0l, b1l, 0,1,2,3,4,5,6,7,8,9,10,11,12,13,14,15);

        // bf16x3: x*w ~= xh*wh + xh*wl + xl*wh
        v8f c = acc[mt * 4 + nt];
        c = __builtin_amdgcn_wmma_f32_16x16x32_bf16(false, ah, false, bh, (short)0, c, false, false);
        c = __builtin_amdgcn_wmma_f32_16x16x32_bf16(false, ah, false, bl, (short)0, c, false, false);
        c = __builtin_amdgcn_wmma_f32_16x16x32_bf16(false, al, false, bh, (short)0, c, false, false);
        acc[mt * 4 + nt] = c;
      }
    }
    __syncthreads();   // all waves done reading `cur` before it is overwritten
  }

  // Write out: fp32, row M = mBase + wm*32 + mt*16 + half*8 + r,
  //                  col N = nBase + wn*64 + nt*16 + l16.
#pragma unroll
  for (int mt = 0; mt < 2; ++mt) {
    const int mRow = mBase + wm * 32 + mt * 16 + half * 8;
#pragma unroll
    for (int nt = 0; nt < 4; ++nt) {
      const int col = nBase + wn * 64 + nt * 16 + l16;
#pragma unroll
      for (int r = 0; r < 8; ++r) {
        out[(size_t)(mRow + r) * N_TOTAL + col] = acc[mt * 4 + nt][r];
      }
    }
  }
}

// ---------------------------------------------------------------------------
// Launch
// ---------------------------------------------------------------------------
extern "C" void kernel_launch(void* const* d_in, const int* in_sizes, int n_in,
                              void* d_out, int out_size, void* d_ws, size_t ws_size,
                              hipStream_t stream) {
  const float* x    = (const float*)d_in[0];  // (4,2048,4096)
  const float* W    = (const float*)d_in[1];  // (4096,4096)
  const float* bias = (const float*)d_in[2];  // (4096)
  const float* ck   = (const float*)d_in[3];  // (16,16,256)
  float* out = (float*)d_out;                 // (4,2048,4096) fp32

  // Workspace carve-up (192 MiB total):
  unsigned short* Xhi = (unsigned short*)d_ws;
  unsigned short* Xlo = Xhi + (size_t)M_TOTAL * K_TOTAL;
  unsigned short* Whi = Xlo + (size_t)M_TOTAL * K_TOTAL;
  unsigned short* Wlo = Whi + (size_t)N_TOTAL * K_TOTAL;

  c3a_split_x<<<(M_TOTAL * (size_t)K_TOTAL) / 256, 256, 0, stream>>>(x, Xhi, Xlo);
  c3a_build_w<<<(N_TOTAL * (size_t)K_TOTAL) / 256, 256, 0, stream>>>(W, ck, Whi, Wlo);

  dim3 grid(N_TOTAL / BN, M_TOTAL / BM);  // (32, 64)
  c3a_gemm<<<grid, 256, 0, stream>>>(Xhi, Xlo, Whi, Wlo, bias, out);
}
